// HMMLayer_13778255086221
// MI455X (gfx1250) — matrix operand order repaired and checked by reading, not compiled, not run on hardware
//
#include <hip/hip_runtime.h>
#include <hip/hip_bf16.h>

#define HMM_B 32
#define HMM_T 4000
#define HMM_K 128
#define HMM_EPS 1e-8f

typedef __attribute__((ext_vector_type(16))) _Float16 v16h;
typedef __attribute__((ext_vector_type(8)))  _Float16 v8h;
typedef __attribute__((ext_vector_type(8)))  float    v8f;

// Raw v_rcp_f32 (~1 ulp); feeds renormalized probabilities, so fine.
__device__ __forceinline__ float fast_rcp(float v) {
  return __builtin_amdgcn_rcpf(v);
}

__device__ __forceinline__ float sigmoidf_fast(float v) {
  return fast_rcp(1.0f + __expf(-v));
}

// A-fragment (16x32 f16) for row-major tile in LDS: lane ln = M row,
// elems 0..15 -> K = 32*kk + 16*g + [0,16).
__device__ __forceinline__ v16h load_afrag(const _Float16* base, int ln, int g, int kk) {
  const v8h lo = *(const v8h*)&base[ln * HMM_K + 32 * kk + 16 * g];
  const v8h hi = *(const v8h*)&base[ln * HMM_K + 32 * kk + 16 * g + 8];
  return __builtin_shufflevector(lo, hi, 0,1,2,3,4,5,6,7,8,9,10,11,12,13,14,15);
}

// ---------------------------------------------------------------------------
// Prep: P = row-softmax(log_transition_logits) -> f16 (plus transposed copy),
//       p0 = softmax(log_initial_logits) -> f32.
// ---------------------------------------------------------------------------
__global__ void hmm_prep_kernel(const float* __restrict__ logT,
                                const float* __restrict__ logPi,
                                _Float16* __restrict__ Pf,
                                _Float16* __restrict__ PTf,
                                float* __restrict__ p0) {
  const int r = threadIdx.x;  // 0..127, one row per thread
  float mx = -3.4e38f;
  for (int c = 0; c < HMM_K; ++c) mx = fmaxf(mx, logT[r * HMM_K + c]);
  float s = 0.f;
  for (int c = 0; c < HMM_K; ++c) s += __expf(logT[r * HMM_K + c] - mx);
  const float inv = 1.0f / s;
  for (int c = 0; c < HMM_K; ++c) {
    const float pv = __expf(logT[r * HMM_K + c] - mx) * inv;
    Pf[r * HMM_K + c]  = (_Float16)pv;   // Pf[k][n]  = P[k][n]
    PTf[c * HMM_K + r] = (_Float16)pv;   // PTf[n][k] = P[k][n]
  }
  float mx2 = -3.4e38f;
  for (int c = 0; c < HMM_K; ++c) mx2 = fmaxf(mx2, logPi[c]);
  float s2 = 0.f;
  for (int c = 0; c < HMM_K; ++c) s2 += __expf(logPi[c] - mx2);
  p0[r] = __expf(logPi[r] - mx2) / s2;
}

// ---------------------------------------------------------------------------
// Backward scan: beta_{t-1} ∝ (b_t * beta_t) @ P^T, stabilized by the
// sum(c_t) normalizer from a ones-WMMA on the same A-fragments (the per-step
// beta scale cancels in the final posterior normalization).
// x loads are software-pipelined one step ahead to hide global latency.
// ---------------------------------------------------------------------------
__global__ __launch_bounds__(256) void hmm_bwd_kernel(
    const float* __restrict__ x, const _Float16* __restrict__ PTf,
    float* __restrict__ out) {
  const int bbase = blockIdx.x * 16;
  const int tid  = threadIdx.x;
  const int wave = tid >> 5;
  const int lane = tid & 31;
  const int g    = lane >> 4;
  const int ln   = lane & 15;
  const int col  = 16 * wave + ln;
  const size_t rowstride = (size_t)HMM_T * HMM_K;

  const float* __restrict__ xblk = x + (size_t)bbase * rowstride;
  float* __restrict__ oblk = out + (size_t)bbase * rowstride;
  unsigned rbase[8], ldsoff[8];
  #pragma unroll
  for (int r = 0; r < 8; ++r) {
    rbase[r]  = (unsigned)((8 * g + r) * (HMM_T * HMM_K)) + (unsigned)col;
    ldsoff[r] = (unsigned)((8 * g + r) * HMM_K + col);
  }

  __shared__ __align__(16) _Float16 cLDS[2][16 * HMM_K];  // c_t = b_t * beta_t

  v16h PTfrag[4];
  #pragma unroll
  for (int kk = 0; kk < 4; ++kk) {
    #pragma unroll
    for (int i = 0; i < 16; ++i)
      PTfrag[kk][i] = PTf[(size_t)(32 * kk + 16 * g + i) * HMM_K + col];
  }
  v16h ONES;
  #pragma unroll
  for (int i = 0; i < 16; ++i) ONES[i] = (_Float16)1.0f;

  // ---- t = T-1: beta = 1, c_{T-1} = sigmoid(x_{T-1}); preload x_{T-2} ----
  float xv[8];
  #pragma unroll
  for (int r = 0; r < 8; ++r) {
    const unsigned off = rbase[r] + (unsigned)((HMM_T - 1) * HMM_K);
    oblk[off] = 1.0f;
    cLDS[1][ldsoff[r]] = (_Float16)sigmoidf_fast(xblk[off]);
    xv[r] = xblk[off - HMM_K];          // x at t = T-2, used at iteration t=T-1
  }
  __syncthreads();

  for (int t = HMM_T - 1; t >= 1; --t) {
    // kick off next iteration's x loads (x at t-2), clamped — no branch
    const unsigned pl = (unsigned)((t >= 2 ? t - 2 : 0) * HMM_K);
    float nxv[8];
    #pragma unroll
    for (int r = 0; r < 8; ++r) nxv[r] = xblk[rbase[r] + pl];

    const _Float16* rbuf = cLDS[t & 1];
    _Float16* wbuf = cLDS[(t & 1) ^ 1];

    v16h cf[4];
    #pragma unroll
    for (int kk = 0; kk < 4; ++kk) cf[kk] = load_afrag(rbuf, ln, g, kk);

    // b_{t-1} from preloaded x (TRANS ops overlap the WMMA chain)
    float sg[8];
    #pragma unroll
    for (int r = 0; r < 8; ++r) sg[r] = sigmoidf_fast(xv[r]);

    v8f acc = {};   // c_t @ P^T
    v8f cs  = {};   // rowsum(c_t) broadcast in every column
    #pragma unroll
    for (int kk = 0; kk < 4; ++kk) {
      acc = __builtin_amdgcn_wmma_f32_16x16x32_f16(false, cf[kk], false, PTfrag[kk],
                                                   (short)0, acc, false, false);
      cs  = __builtin_amdgcn_wmma_f32_16x16x32_f16(false, cf[kk], false, ONES,
                                                   (short)0, cs, false, false);
    }
    const unsigned toff = (unsigned)((t - 1) * HMM_K);
    #pragma unroll
    for (int r = 0; r < 8; ++r) {
      const float beta = acc[r] * fast_rcp(cs[r] + HMM_EPS);
      oblk[rbase[r] + toff] = beta;
      wbuf[ldsoff[r]] = (_Float16)(sg[r] * beta);
    }
    #pragma unroll
    for (int r = 0; r < 8; ++r) xv[r] = nxv[r];
    __syncthreads();
  }
}

// ---------------------------------------------------------------------------
// Forward scan with deferred normalization:
//   u_t = (u_{t-1} @ P) * rcp(sum(u_{t-1})+eps) * b_t   (u stored f16 in LDS)
// Writes the unnormalized posterior numerator u_t * beta_t in place over beta.
// x and beta loads are software-pipelined one step ahead.
// ---------------------------------------------------------------------------
__global__ __launch_bounds__(256) void hmm_fwd_kernel(
    const float* __restrict__ x, const _Float16* __restrict__ Pf,
    const float* __restrict__ p0, float* __restrict__ out) {
  const int bbase = blockIdx.x * 16;
  const int tid  = threadIdx.x;
  const int wave = tid >> 5;
  const int lane = tid & 31;
  const int g    = lane >> 4;
  const int ln   = lane & 15;
  const int col  = 16 * wave + ln;
  const size_t rowstride = (size_t)HMM_T * HMM_K;

  const float* __restrict__ xblk = x + (size_t)bbase * rowstride;
  float* __restrict__ oblk = out + (size_t)bbase * rowstride;
  unsigned rbase[8], ldsoff[8];
  #pragma unroll
  for (int r = 0; r < 8; ++r) {
    rbase[r]  = (unsigned)((8 * g + r) * (HMM_T * HMM_K)) + (unsigned)col;
    ldsoff[r] = (unsigned)((8 * g + r) * HMM_K + col);
  }

  __shared__ __align__(16) _Float16 aLDS[2][16 * HMM_K];  // unnormalized alpha

  v16h Pfrag[4];
  #pragma unroll
  for (int kk = 0; kk < 4; ++kk) {
    #pragma unroll
    for (int i = 0; i < 16; ++i)
      Pfrag[kk][i] = Pf[(size_t)(32 * kk + 16 * g + i) * HMM_K + col];
  }
  v16h ONES;
  #pragma unroll
  for (int i = 0; i < 16; ++i) ONES[i] = (_Float16)1.0f;

  // ---- t = 0: u_0 = p0 * b_0; numerator u_0 * beta_0; preload t=1 ----
  float xv[8], bv[8];
  #pragma unroll
  for (int r = 0; r < 8; ++r) {
    const unsigned off = rbase[r];
    const float u = p0[col] * sigmoidf_fast(xblk[off]);
    aLDS[0][ldsoff[r]] = (_Float16)u;
    oblk[off] = u * oblk[off];          // out held beta_0
    xv[r] = xblk[off + HMM_K];          // x at t=1
    bv[r] = oblk[off + HMM_K];          // beta at t=1
  }
  __syncthreads();

  for (int t = 1; t < HMM_T; ++t) {
    // kick off next iteration's x/beta loads (clamped at the end — no branch)
    const unsigned pl = (unsigned)((t + 1 < HMM_T ? t + 1 : t) * HMM_K);
    float nxv[8], nbv[8];
    #pragma unroll
    for (int r = 0; r < 8; ++r) {
      nxv[r] = xblk[rbase[r] + pl];
      nbv[r] = oblk[rbase[r] + pl];
    }

    const _Float16* rbuf = aLDS[(t - 1) & 1];
    _Float16* wbuf = aLDS[t & 1];

    v16h af[4];
    #pragma unroll
    for (int kk = 0; kk < 4; ++kk) af[kk] = load_afrag(rbuf, ln, g, kk);

    // emissions from preloaded x (TRANS ops overlap the WMMA chain)
    float sg[8];
    #pragma unroll
    for (int r = 0; r < 8; ++r) sg[r] = sigmoidf_fast(xv[r]);

    v8f acc = {};   // u_{t-1} @ P
    v8f us  = {};   // rowsum(u_{t-1}) broadcast in every column
    #pragma unroll
    for (int kk = 0; kk < 4; ++kk) {
      acc = __builtin_amdgcn_wmma_f32_16x16x32_f16(false, af[kk], false, Pfrag[kk],
                                                   (short)0, acc, false, false);
      us  = __builtin_amdgcn_wmma_f32_16x16x32_f16(false, af[kk], false, ONES,
                                                   (short)0, us, false, false);
    }
    const unsigned toff = (unsigned)(t * HMM_K);
    #pragma unroll
    for (int r = 0; r < 8; ++r) {
      const float u = acc[r] * fast_rcp(us[r] + HMM_EPS) * sg[r];
      wbuf[ldsoff[r]] = (_Float16)u;
      oblk[rbase[r] + toff] = u * bv[r];   // posterior numerator over beta_t
    }
    #pragma unroll
    for (int r = 0; r < 8; ++r) { xv[r] = nxv[r]; bv[r] = nbv[r]; }
    __syncthreads();
  }
}

// ---------------------------------------------------------------------------
// Posterior row-normalization: out[row, :] /= (sum(out[row, :]) + eps).
// 128,000 independent rows; one wave per row; bandwidth bound.
// ---------------------------------------------------------------------------
__global__ __launch_bounds__(256) void hmm_pnorm_kernel(float* __restrict__ out) {
  const size_t row = (size_t)blockIdx.x * 8 + (threadIdx.x >> 5);
  const int lane = threadIdx.x & 31;
  float4* p = (float4*)(out + row * HMM_K) + lane;
  float4 v = *p;
  float s = v.x + v.y + v.z + v.w;
  s += __shfl_xor(s, 1, 32);
  s += __shfl_xor(s, 2, 32);
  s += __shfl_xor(s, 4, 32);
  s += __shfl_xor(s, 8, 32);
  s += __shfl_xor(s, 16, 32);
  const float inv = fast_rcp(s + HMM_EPS);
  v.x *= inv; v.y *= inv; v.z *= inv; v.w *= inv;
  *p = v;
}

// ---------------------------------------------------------------------------
extern "C" void kernel_launch(void* const* d_in, const int* in_sizes, int n_in,
                              void* d_out, int out_size, void* d_ws, size_t ws_size,
                              hipStream_t stream) {
  const float* x     = (const float*)d_in[0];  // (32, 4000, 128) f32
  const float* logT  = (const float*)d_in[1];  // (128, 128) f32
  const float* logPi = (const float*)d_in[2];  // (128,) f32
  float* out = (float*)d_out;                  // (32, 4000, 128) f32

  _Float16* Pf  = (_Float16*)d_ws;             // 32 KB
  _Float16* PTf = Pf + HMM_K * HMM_K;          // 32 KB
  float*    p0  = (float*)(PTf + HMM_K * HMM_K);

  hipLaunchKernelGGL(hmm_prep_kernel, dim3(1), dim3(HMM_K), 0, stream,
                     logT, logPi, Pf, PTf, p0);
  hipLaunchKernelGGL(hmm_bwd_kernel, dim3(HMM_B / 16), dim3(256), 0, stream,
                     x, PTf, out);
  hipLaunchKernelGGL(hmm_fwd_kernel, dim3(HMM_B / 16), dim3(256), 0, stream,
                     x, Pf, p0, out);
  hipLaunchKernelGGL(hmm_pnorm_kernel, dim3((HMM_B * HMM_T) / 8), dim3(256), 0, stream,
                     out);
}
